// CorrLookupX_53051436040752
// MI455X (gfx1250) — compile-verified, matrix operand order checked by reading, not run on hardware
//
#include <hip/hip_runtime.h>
#include <hip/hip_bf16.h>

typedef __attribute__((ext_vector_type(16))) __bf16 v16bf;
typedef __attribute__((ext_vector_type(8)))  __bf16 v8bf;
typedef __attribute__((ext_vector_type(8)))  float  v8f;

#define HH 80
#define WW 80
#define PP 6400           // H*W
#define CC 256            // input channels
#define CH 324            // C_HALF
#define CP 352            // padded channels (multiple of 32, and of 8)
#define NPT 81            // points per level

__device__ __forceinline__ unsigned short f2bfu(float f) {
    union { float f; unsigned u; } v; v.f = f;
    return (unsigned short)((v.u + 0x7FFFu + ((v.u >> 16) & 1u)) >> 16);
}
__device__ __forceinline__ __bf16 f2bf(float f) {
    union { unsigned short s; __bf16 b; } o; o.s = f2bfu(f);
    return o.b;
}

// k -> position inside the fragment-ready 32-entry per-column LDS run.
// Lane kgroup 0 reads positions [0,16), kgroup 1 reads [16,32), each as 2x b128.
__device__ __forceinline__ int kpos(int k) {
    return ((k >> 3) & 1) * 16 + ((k >> 4) & 1) * 8 + (k & 7);
}

// two contiguous 16B LDS chunks -> v16bf fragment
__device__ __forceinline__ v16bf frag2(const unsigned short* p) {
    v8bf lo = *(const v8bf*)p;
    v8bf hi = *(const v8bf*)(p + 8);
    return __builtin_shufflevector(lo, hi, 0, 1, 2, 3, 4, 5, 6, 7,
                                   8, 9, 10, 11, 12, 13, 14, 15);
}
// A fragment: row-major tile, runs at [kg, kg+8) and [16+kg, 16+kg+8)
__device__ __forceinline__ v16bf fragA(const unsigned short* As, int row, int kg) {
    v8bf lo = *(const v8bf*)(As + row * 32 + kg);
    v8bf hi = *(const v8bf*)(As + row * 32 + 16 + kg);
    return __builtin_shufflevector(lo, hi, 0, 1, 2, 3, 4, 5, 6, 7,
                                   8, 9, 10, 11, 12, 13, 14, 15);
}

// ---------------------------------------------------------------------------
// 1. pack f0 -> A (P x C, bf16) transposed, f1 -> B (C x P, bf16)
// ---------------------------------------------------------------------------
__global__ void pack_inputs(const float* __restrict__ f0, const float* __restrict__ f1,
                            __bf16* __restrict__ A, __bf16* __restrict__ Bm) {
    int idx = blockIdx.x * blockDim.x + threadIdx.x;
    if (idx >= CC * PP) return;
    int c = idx / PP, p = idx % PP;
    A[(size_t)p * CC + c] = f2bf(f0[idx]);
    Bm[idx]              = f2bf(f1[idx]);
}

// ---------------------------------------------------------------------------
// 2. pack conv weights (3,3,324,324) HWIO -> (9, 352, 352) bf16, zero padded
// ---------------------------------------------------------------------------
__global__ void pack_weights(const float* __restrict__ w, __bf16* __restrict__ wp) {
    int idx = blockIdx.x * blockDim.x + threadIdx.x;
    if (idx >= 9 * CP * CP) return;
    int tap = idx / (CP * CP);
    int r   = (idx / CP) % CP;     // ci
    int c   = idx % CP;            // co
    float v = (r < CH && c < CH) ? w[((size_t)tap * CH + r) * CH + c] : 0.0f;
    wp[idx] = f2bf(v);
}

// ---------------------------------------------------------------------------
// 3. Correlation GEMM: 128x128 tile / block, 8 waves, bf16 WMMA 16x16x32.
//    A staged row-major (b128 in/out); B staged fragment-ready transposed.
// ---------------------------------------------------------------------------
__global__ void __launch_bounds__(256)
gemm_corr(const __bf16* __restrict__ A, const __bf16* __restrict__ Bm,
          float* __restrict__ corr, float* __restrict__ corrT) {
    __shared__ __align__(16) unsigned short AsU[128 * 32];
    __shared__ __align__(16) unsigned short Bs2[128 * 32];

    const int tid   = threadIdx.x;
    const int lane  = tid & 31;
    const int wv    = tid >> 5;          // 0..7
    const int pBase = blockIdx.x * 128;
    const int qBase = blockIdx.y * 128;

    v8f acc[8];
#pragma unroll
    for (int j = 0; j < 8; ++j) acc[j] = (v8f)(0.0f);

    const int mrow  = wv * 16 + (lane & 15);
    const int kg    = (lane >> 4) << 3;     // 0 or 8
    const int kgsel = (lane >> 4) << 4;     // 0 or 16 (fragment-ready B offset)

    // B cooperative-load mapping: thread handles k-pair (2*kp, 2*kp+1), 8 n's
    const int kp  = tid >> 4;               // 0..15
    const int nb  = (tid & 15) * 8;         // 0..120
    const int bp  = kpos(2 * kp);           // even position, pair-contiguous

    for (int kk = 0; kk < 8; ++kk) {
        __syncthreads();
        // A tile 128x32: 512 uint4, 2 per thread (global_load_b128 -> ds b128)
#pragma unroll
        for (int i = 0; i < 2; ++i) {
            int e = tid + i * 256;
            int r = e >> 2, cchunk = (e & 3) * 8;
            *(uint4*)(AsU + r * 32 + cchunk) =
                *(const uint4*)(A + (size_t)(pBase + r) * CC + kk * 32 + cchunk);
        }
        // B tile 32x128, transposed into fragment-ready layout.
        {
            const __bf16* g0 = Bm + (size_t)(kk * 32 + 2 * kp) * PP + qBase + nb;
            uint4 ra = *(const uint4*)g0;
            uint4 rb = *(const uint4*)(g0 + PP);
            const unsigned short* ua = (const unsigned short*)&ra;
            const unsigned short* ub = (const unsigned short*)&rb;
#pragma unroll
            for (int i = 0; i < 8; ++i) {
                unsigned v = (unsigned)ua[i] | ((unsigned)ub[i] << 16);
                *(unsigned*)(Bs2 + (nb + i) * 32 + bp) = v;
            }
        }
        if (kk < 7) {
            __builtin_prefetch(&A[(size_t)(pBase + (tid >> 1)) * CC + (kk + 1) * 32], 0, 1);
            __builtin_prefetch(&Bm[(size_t)((kk + 1) * 32 + kp) * PP + qBase + nb], 0, 1);
        }
        __syncthreads();

        v16bf af = fragA(AsU, mrow, kg);        // 2x ds_load_b128
#pragma unroll
        for (int j = 0; j < 8; ++j) {
            v16bf bf_ = frag2(Bs2 + (j * 16 + (lane & 15)) * 32 + kgsel);
            acc[j] = __builtin_amdgcn_wmma_f32_16x16x32_bf16(
                false, af, false, bf_, (short)0, acc[j], false, false);
        }
    }

    // C/D layout: VGPR r -> M = r (+8 for lanes 16..31), N = lane&15
#pragma unroll
    for (int j = 0; j < 8; ++j) {
#pragma unroll
        for (int r = 0; r < 8; ++r) {
            int row = pBase + wv * 16 + r + ((lane >> 4) << 3);
            int col = qBase + j * 16 + (lane & 15);
            float v = acc[j][r];
            corr [(size_t)row * PP + col] = v;
            corrT[(size_t)col * PP + row] = v;
        }
    }
}

// ---------------------------------------------------------------------------
// 4. 2x2 avg pool over trailing (h, w) of an (N, h, w) tensor
// ---------------------------------------------------------------------------
__global__ void avgpool2x2(const float* __restrict__ in, float* __restrict__ out,
                           int n, int h, int w) {
    int oh = h >> 1, ow = w >> 1;
    long long total = (long long)n * oh * ow;
    long long idx = (long long)blockIdx.x * blockDim.x + threadIdx.x;
    if (idx >= total) return;
    int q  = (int)(idx / (oh * ow));
    int rr = (int)(idx % (oh * ow));
    int oy = rr / ow, ox = rr % ow;
    const float* b = in + (size_t)q * h * w;
    out[idx] = 0.25f * (b[(2 * oy) * w + 2 * ox]     + b[(2 * oy) * w + 2 * ox + 1] +
                        b[(2 * oy + 1) * w + 2 * ox] + b[(2 * oy + 1) * w + 2 * ox + 1]);
}

// ---------------------------------------------------------------------------
// 5. Pyramid lookup: 81 bilinear taps / level, zero padding outside.
// ---------------------------------------------------------------------------
__device__ __forceinline__ float bilin(const float* __restrict__ base, int hl, int wl,
                                       float x, float y) {
    float x0f = floorf(x), y0f = floorf(y);
    int x0 = (int)x0f, y0 = (int)y0f;
    float wx1 = x - x0f, wy1 = y - y0f;
    float acc = 0.0f;
#pragma unroll
    for (int dy = 0; dy < 2; ++dy) {
#pragma unroll
        for (int dx = 0; dx < 2; ++dx) {
            int xi = x0 + dx, yi = y0 + dy;
            float wgt = (dx ? wx1 : 1.0f - wx1) * (dy ? wy1 : 1.0f - wy1);
            if (xi >= 0 && xi < wl && yi >= 0 && yi < hl)
                acc += base[yi * wl + xi] * wgt;
        }
    }
    return acc;
}

__global__ void lookup_pyr(const float* __restrict__ L0, const float* __restrict__ L1,
                           const float* __restrict__ L2, const float* __restrict__ L3,
                           float* __restrict__ feat) {
    int idx = blockIdx.x * blockDim.x + threadIdx.x;
    if (idx >= PP * CH) return;
    int p   = idx / CH;
    int ch  = idx % CH;
    int lvl = ch / NPT;
    int pi  = ch % NPT;
    int px = pi / 9 - 4;      // points sorted by (x, y): x-major
    int py = pi % 9 - 4;
    int s  = 1 << lvl;
    int hl = HH >> lvl, wl = WW >> lvl;
    int i = p / WW, j = p % WW;
    float x = (2.0f * j + 1.0f) / (2.0f * s) - 0.5f + (float)px;
    float y = (2.0f * i + 1.0f) / (2.0f * s) - 0.5f + (float)py;
    const float* base = (lvl == 0 ? L0 : lvl == 1 ? L1 : lvl == 2 ? L2 : L3)
                        + (size_t)p * hl * wl;
    feat[(size_t)p * CH + ch] = bilin(base, hl, wl, x, y);
}

// ---------------------------------------------------------------------------
// 6. feat (P,324) f32 -> (P,352) bf16, zero padded
// ---------------------------------------------------------------------------
__global__ void pack_feat(const float* __restrict__ feat, __bf16* __restrict__ fb) {
    int idx = blockIdx.x * blockDim.x + threadIdx.x;
    if (idx >= PP * CP) return;
    int p = idx / CP, c = idx % CP;
    fb[idx] = f2bf(c < CH ? feat[(size_t)p * CH + c] : 0.0f);
}

// ---------------------------------------------------------------------------
// 7. 3x3 conv, implicit-GEMM WMMA. Tile: 64 pixels x 128 out channels,
//    8 waves as 4(m) x 2(n), 4 WMMAs / wave / K-step. K = 9 taps x 352.
//    mode 1: relu -> bf16 act (padded). mode 2: resid + relu -> f32 y.
// ---------------------------------------------------------------------------
__global__ void __launch_bounds__(256)
conv3x3_wmma(const __bf16* __restrict__ in, const __bf16* __restrict__ wp,
             const float* __restrict__ bias, const float* __restrict__ resid,
             float* __restrict__ outf, __bf16* __restrict__ outbf, int mode) {
    __shared__ __align__(16) unsigned short AsU[64 * 32];
    __shared__ __align__(16) unsigned short Bs2[128 * 32];

    const int tid  = threadIdx.x;
    const int lane = tid & 31;
    const int wv   = tid >> 5;
    const int pB   = blockIdx.x * 64;    // pixel base
    const int nB   = blockIdx.y * 128;   // out-channel base
    const int mT   = (wv & 3) * 16;      // wave's m-tile
    const int n0   = (wv >> 2) * 64;     // wave's n-group (4 tiles of 16)

    v8f acc[4];
#pragma unroll
    for (int j = 0; j < 4; ++j) acc[j] = (v8f)(0.0f);

    const int kg    = (lane >> 4) << 3;
    const int kgsel = (lane >> 4) << 4;

    const int kp = tid >> 4;             // B-load k pair
    const int nb = (tid & 15) * 8;       // B-load n chunk
    const int bp = kpos(2 * kp);

    // A-load mapping: 64 rows x 32 cols = 256 uint4, one per thread
    const int ar = tid >> 2;
    const int ac = (tid & 3) * 8;
    const int apy = (pB + ar) / WW, apx = (pB + ar) % WW;

    for (int tap = 0; tap < 9; ++tap) {
        int ky = tap / 3 - 1, kx = tap % 3 - 1;
        int iy = apy + ky, ix = apx + kx;
        bool aok = (iy >= 0 && iy < HH && ix >= 0 && ix < WW);
        const __bf16* abase = in + (size_t)(iy * WW + ix) * CP;
        for (int ks = 0; ks < CP / 32; ++ks) {
            __syncthreads();
            // A tile 64x32
            {
                uint4 v = {0u, 0u, 0u, 0u};
                if (aok) v = *(const uint4*)(abase + ks * 32 + ac);
                *(uint4*)(AsU + ar * 32 + ac) = v;
            }
            // B tile 32x128 (weights), fragment-ready transposed
            {
                int co = nB + nb;
                uint4 ra = {0u, 0u, 0u, 0u}, rb = {0u, 0u, 0u, 0u};
                if (co < CP) {
                    const __bf16* g0 =
                        wp + ((size_t)tap * CP + ks * 32 + 2 * kp) * CP + co;
                    ra = *(const uint4*)g0;
                    rb = *(const uint4*)(g0 + CP);
                }
                const unsigned short* ua = (const unsigned short*)&ra;
                const unsigned short* ub = (const unsigned short*)&rb;
#pragma unroll
                for (int i = 0; i < 8; ++i) {
                    unsigned v = (unsigned)ua[i] | ((unsigned)ub[i] << 16);
                    *(unsigned*)(Bs2 + (nb + i) * 32 + bp) = v;
                }
            }
            __syncthreads();

            v16bf af = fragA(AsU, mT + (lane & 15), kg);
#pragma unroll
            for (int j = 0; j < 4; ++j) {
                v16bf bf_ = frag2(Bs2 + (n0 + j * 16 + (lane & 15)) * 32 + kgsel);
                acc[j] = __builtin_amdgcn_wmma_f32_16x16x32_bf16(
                    false, af, false, bf_, (short)0, acc[j], false, false);
            }
        }
    }

    // Epilogue
#pragma unroll
    for (int j = 0; j < 4; ++j) {
#pragma unroll
        for (int r = 0; r < 8; ++r) {
            int p  = pB + mT + r + ((lane >> 4) << 3);
            int co = nB + n0 + j * 16 + (lane & 15);
            float b = (co < CH) ? bias[co] : 0.0f;
            float v = acc[j][r] + b;
            v = v > 0.0f ? v : 0.0f;
            if (mode == 1) {
                if (co < CP) outbf[(size_t)p * CP + co] = f2bf(v);
            } else {
                if (co < CH) outf[(size_t)p * CH + co] = resid[(size_t)p * CH + co] + v;
            }
        }
    }
}

// ---------------------------------------------------------------------------
// 8. LayerNorm over 324 channels + transpose to NCHW output slice
// ---------------------------------------------------------------------------
__global__ void ln_writeout(const float* __restrict__ y, const float* __restrict__ g,
                            const float* __restrict__ b, float* __restrict__ out,
                            int branch) {
    __shared__ float s1[128], s2[128];
    int p = blockIdx.x;
    int tid = threadIdx.x;
    float sum = 0.0f, sq = 0.0f;
    for (int c = tid; c < CH; c += 128) {
        float v = y[(size_t)p * CH + c];
        sum += v; sq += v * v;
    }
    s1[tid] = sum; s2[tid] = sq;
    __syncthreads();
    for (int s = 64; s > 0; s >>= 1) {
        if (tid < s) { s1[tid] += s1[tid + s]; s2[tid] += s2[tid + s]; }
        __syncthreads();
    }
    float mean = s1[0] / (float)CH;
    float var  = s2[0] / (float)CH - mean * mean;
    float inv  = rsqrtf(var + 1e-5f);
    for (int c = tid; c < CH; c += 128) {
        float v = (y[(size_t)p * CH + c] - mean) * inv * g[c] + b[c];
        out[((size_t)(branch * CH + c)) * PP + p] = v;
    }
}

// ---------------------------------------------------------------------------
// Host launcher
// ---------------------------------------------------------------------------
extern "C" void kernel_launch(void* const* d_in, const int* in_sizes, int n_in,
                              void* d_out, int out_size, void* d_ws, size_t ws_size,
                              hipStream_t stream) {
    const float* f0  = (const float*)d_in[0];
    const float* f1  = (const float*)d_in[1];
    const float* w1  = (const float*)d_in[2];
    const float* b1  = (const float*)d_in[3];
    const float* w2  = (const float*)d_in[4];
    const float* b2  = (const float*)d_in[5];
    const float* lng = (const float*)d_in[6];
    const float* lnb = (const float*)d_in[7];
    float* out = (float*)d_out;

    char* ws = (char*)d_ws;
    size_t off = 0;
    auto take = [&](size_t bytes) -> void* {
        void* p = ws + off;
        off += (bytes + 255) & ~(size_t)255;
        return p;
    };

    const size_t PPsq = (size_t)PP * PP;
    float*  corr  = (float*)take(PPsq * 4);
    float*  corrT = (float*)take(PPsq * 4);
    float*  pyr0[3], *pyr1[3];
    for (int l = 0; l < 3; ++l) pyr0[l] = (float*)take((size_t)PP * (HH >> (l+1)) * (WW >> (l+1)) * 4);
    for (int l = 0; l < 3; ++l) pyr1[l] = (float*)take((size_t)PP * (HH >> (l+1)) * (WW >> (l+1)) * 4);
    __bf16* Abf  = (__bf16*)take((size_t)PP * CC * 2);
    __bf16* Bbf  = (__bf16*)take((size_t)CC * PP * 2);
    float*  feat0 = (float*)take((size_t)PP * CH * 4);
    float*  feat1 = (float*)take((size_t)PP * CH * 4);
    __bf16* fbf0 = (__bf16*)take((size_t)PP * CP * 2);
    __bf16* fbf1 = (__bf16*)take((size_t)PP * CP * 2);
    __bf16* act0 = (__bf16*)take((size_t)PP * CP * 2);
    __bf16* act1 = (__bf16*)take((size_t)PP * CP * 2);
    float*  y0   = (float*)take((size_t)PP * CH * 4);
    float*  y1   = (float*)take((size_t)PP * CH * 4);
    __bf16* w1bf = (__bf16*)take((size_t)9 * CP * CP * 2);
    __bf16* w2bf = (__bf16*)take((size_t)9 * CP * CP * 2);

    pack_inputs<<<(CC * PP + 255) / 256, 256, 0, stream>>>(f0, f1, Abf, Bbf);
    pack_weights<<<(9 * CP * CP + 255) / 256, 256, 0, stream>>>(w1, w1bf);
    pack_weights<<<(9 * CP * CP + 255) / 256, 256, 0, stream>>>(w2, w2bf);

    gemm_corr<<<dim3(PP / 128, PP / 128), 256, 0, stream>>>(Abf, Bbf, corr, corrT);

    {
        long long t1 = (long long)PP * 40 * 40, t2 = (long long)PP * 20 * 20, t3 = (long long)PP * 10 * 10;
        avgpool2x2<<<(unsigned)((t1 + 255) / 256), 256, 0, stream>>>(corr,    pyr0[0], PP, 80, 80);
        avgpool2x2<<<(unsigned)((t2 + 255) / 256), 256, 0, stream>>>(pyr0[0], pyr0[1], PP, 40, 40);
        avgpool2x2<<<(unsigned)((t3 + 255) / 256), 256, 0, stream>>>(pyr0[1], pyr0[2], PP, 20, 20);
        avgpool2x2<<<(unsigned)((t1 + 255) / 256), 256, 0, stream>>>(corrT,   pyr1[0], PP, 80, 80);
        avgpool2x2<<<(unsigned)((t2 + 255) / 256), 256, 0, stream>>>(pyr1[0], pyr1[1], PP, 40, 40);
        avgpool2x2<<<(unsigned)((t3 + 255) / 256), 256, 0, stream>>>(pyr1[1], pyr1[2], PP, 20, 20);
    }

    lookup_pyr<<<(PP * CH + 255) / 256, 256, 0, stream>>>(corr,  pyr0[0], pyr0[1], pyr0[2], feat0);
    lookup_pyr<<<(PP * CH + 255) / 256, 256, 0, stream>>>(corrT, pyr1[0], pyr1[1], pyr1[2], feat1);

    pack_feat<<<(PP * CP + 255) / 256, 256, 0, stream>>>(feat0, fbf0);
    pack_feat<<<(PP * CP + 255) / 256, 256, 0, stream>>>(feat1, fbf1);

    dim3 cgrid(PP / 64, 3);
    conv3x3_wmma<<<cgrid, 256, 0, stream>>>(fbf0, w1bf, b1, nullptr, nullptr, act0, 1);
    conv3x3_wmma<<<cgrid, 256, 0, stream>>>(act0, w2bf, b2, feat0, y0, nullptr, 2);
    conv3x3_wmma<<<cgrid, 256, 0, stream>>>(fbf1, w1bf, b1, nullptr, nullptr, act1, 1);
    conv3x3_wmma<<<cgrid, 256, 0, stream>>>(act1, w2bf, b2, feat1, y1, nullptr, 2);

    ln_writeout<<<PP, 128, 0, stream>>>(y0, lng, lnb, out, 0);
    ln_writeout<<<PP, 128, 0, stream>>>(y1, lng, lnb, out, 1);
}
